// SSDResnet_61486751809854
// MI455X (gfx1250) — compile-verified, matrix operand order checked by reading, not run on hardware
//
#include <hip/hip_runtime.h>

typedef __attribute__((ext_vector_type(16))) _Float16 v16h;
typedef __attribute__((ext_vector_type(8)))  _Float16 v8h;
typedef __attribute__((ext_vector_type(8)))  float    v8f;
typedef __attribute__((ext_vector_type(4)))  float    v4f;
typedef __attribute__((ext_vector_type(2)))  float    v2f;

#define NEG_SLOPE 0.2f

// ---------------------------------------------------------------------------
// Small GEMM: C[M,N] = op(A[M,Kd] @ B[Kd,N]); op = leaky-relu if leaky != 0.
// Requirements (all call sites satisfy): M % 16 == 0, N % 128 == 0, Kd % 2 == 0.
// fp32 inputs staged to LDS as f16; fp32 accumulate via v_wmma_f32_16x16x32_f16.
// block = 256 threads (8 waves); wave w owns the 16x16 tile at cols w*16.
// grid  = (N/128, M/16).
// B tile is stored TRANSPOSED in LDS so both A and B fragments are two
// contiguous 16-byte runs per lane -> ds_load_b128 pairs, no u16 gathers.
// ---------------------------------------------------------------------------
__global__ __launch_bounds__(256)
void gemm_wmma_f16(const float* __restrict__ A, const float* __restrict__ B,
                   float* __restrict__ C, int M, int Kd, int N, int leaky)
{
    __shared__ __align__(16) _Float16 lA [ 16][40];   // A tile  [m][k], pitch 80 B
    __shared__ __align__(16) _Float16 lBt[128][40];   // B tile^T [n][k], pitch 80 B

    const int tid   = threadIdx.x;
    const int wave  = tid >> 5;
    const int lane  = tid & 31;
    const int half  = lane >> 4;       // 0: lanes 0-15, 1: lanes 16-31
    const int l16   = lane & 15;
    const int kbase = half * 8;        // per documented 16-bit fragment layout

    const int m0 = blockIdx.y * 16;
    const int n0 = blockIdx.x * 128;
    const int nw = wave * 16;

    // A staging: thread -> (row ar, 2 contiguous k at ak)
    const int ar = tid >> 4;           // 0..15
    const int ak = (tid & 15) * 2;     // 0,2,..,30

    v8f acc = {};

    for (int k0 = 0; k0 < Kd; k0 += 32) {
        const int kRem = Kd - k0;

        if (kRem >= 32) {
            // ---- fast path: no guards, vectorized global loads ----
            v2f av = *(const v2f*)(A + (size_t)(m0 + ar) * Kd + k0 + ak);
            lA[ar][ak]     = (_Float16)av.x;
            lA[ar][ak + 1] = (_Float16)av.y;

            #pragma unroll
            for (int q = tid; q < 1024; q += 256) {
                int r = q >> 5, c4 = (q & 31) * 4;           // r: k-row, c4: n
                v4f v = *(const v4f*)(B + (size_t)(k0 + r) * N + n0 + c4);
                lBt[c4 + 0][r] = (_Float16)v.x;
                lBt[c4 + 1][r] = (_Float16)v.y;
                lBt[c4 + 2][r] = (_Float16)v.z;
                lBt[c4 + 3][r] = (_Float16)v.w;
            }
        } else {
            // ---- K tail (only final k-step of Kd=300 / Kd=80): zero-pad ----
            #pragma unroll
            for (int j = 0; j < 2; ++j) {
                int k = ak + j;
                float v = (k < kRem) ? A[(size_t)(m0 + ar) * Kd + k0 + k] : 0.f;
                lA[ar][k] = (_Float16)v;
            }
            #pragma unroll
            for (int q = tid; q < 1024; q += 256) {
                int r = q >> 5, c4 = (q & 31) * 4;
                v4f v = {0.f, 0.f, 0.f, 0.f};
                if (r < kRem)                                  // wave-uniform
                    v = *(const v4f*)(B + (size_t)(k0 + r) * N + n0 + c4);
                lBt[c4 + 0][r] = (_Float16)v.x;
                lBt[c4 + 1][r] = (_Float16)v.y;
                lBt[c4 + 2][r] = (_Float16)v.z;
                lBt[c4 + 3][r] = (_Float16)v.w;
            }
        }
        __syncthreads();

        // fragments: two 16-B contiguous runs each -> ds_load_b128 pairs
        v8h alo = *(const v8h*)&lA [l16]      [kbase];        // K kbase..+7
        v8h ahi = *(const v8h*)&lA [l16]      [kbase + 16];   // K kbase+16..+23
        v8h blo = *(const v8h*)&lBt[nw + l16] [kbase];
        v8h bhi = *(const v8h*)&lBt[nw + l16] [kbase + 16];
        v16h af = __builtin_shufflevector(alo, ahi,
                    0,1,2,3,4,5,6,7,8,9,10,11,12,13,14,15);
        v16h bf = __builtin_shufflevector(blo, bhi,
                    0,1,2,3,4,5,6,7,8,9,10,11,12,13,14,15);

        acc = __builtin_amdgcn_wmma_f32_16x16x32_f16(
                  /*neg_a=*/false, af, /*neg_b=*/false, bf,
                  /*c_mod=*/(short)0, acc, /*reuse_a=*/false, /*reuse_b=*/false);
        __syncthreads();
    }

    // C/D layout: VGPR r -> M = half*8 + r, N = l16 (no guards: exact tiling)
    const int n = n0 + nw + l16;
    #pragma unroll
    for (int r = 0; r < 8; ++r) {
        float v = acc[r];
        if (leaky) v = (v > 0.f) ? v : NEG_SLOPE * v;
        C[(size_t)(m0 + half * 8 + r) * N + n] = v;
    }
}

// ---------------------------------------------------------------------------
// Phase 2: out[b, branch, k, i] = sum_{kk<5} feature[b,i+kk] * x[k,i+kk]
// One block per (b, k, branch). Product row cached in LDS; 4 outputs from two
// ds_load_b128 reads; 16-B nontemporal store (335 MB write-once stream: keep
// it from evicting the L2-resident feature/x rows).
// ---------------------------------------------------------------------------
__global__ __launch_bounds__(256)
void winsum_kernel(const float* __restrict__ feature,
                   const float* __restrict__ xloc,
                   const float* __restrict__ xglob,
                   float* __restrict__ out)
{
    __shared__ __align__(16) float p[2048];

    const int tid    = threadIdx.x;
    const int bk     = blockIdx.x;            // 0..159
    const int branch = bk / 80;
    const int k      = bk - branch * 80;
    const int b      = blockIdx.y;

    const float* __restrict__ f = feature + (size_t)b * 2048;
    const float* __restrict__ x = (branch ? xglob : xloc) + (size_t)k * 2048;

    #pragma unroll
    for (int c = tid; c < 2048; c += 256) p[c] = f[c] * x[c];
    __syncthreads();

    float* orow = out + (size_t)b * 327040 + (size_t)branch * 163520
                      + (size_t)k * 2044;

    for (int idx = tid; idx < 511; idx += 256) {   // 511*4 = 2044 outputs
        int i = idx * 4;
        v4f a  = *(const v4f*)(p + i);
        v4f bb = *(const v4f*)(p + i + 4);
        v4f o;
        o.x = a.x + a.y + a.z + a.w + bb.x;
        o.y = a.y + a.z + a.w + bb.x + bb.y;
        o.z = a.z + a.w + bb.x + bb.y + bb.z;
        o.w = a.w + bb.x + bb.y + bb.z + bb.w;
        __builtin_nontemporal_store(o, (v4f*)(orow + i));
    }
}

// ---------------------------------------------------------------------------
extern "C" void kernel_launch(void* const* d_in, const int* in_sizes, int n_in,
                              void* d_out, int out_size, void* d_ws, size_t ws_size,
                              hipStream_t stream)
{
    (void)in_sizes; (void)n_in; (void)out_size; (void)ws_size;

    const float* feature = (const float*)d_in[0];   // [256,2048]
    const float* inp     = (const float*)d_in[1];   // [1,80,300]
    const float* cor     = (const float*)d_in[2];   // [1,80,1536]
    const float* W1      = (const float*)d_in[3];   // [300,1024]
    const float* W2      = (const float*)d_in[4];   // [1024,2048]
    const float* W3      = (const float*)d_in[5];   // [1536,4096]
    const float* W4      = (const float*)d_in[6];   // [4096,2048]
    const float* A1      = (const float*)d_in[7];   // [80,80]
    const float* A2      = (const float*)d_in[8];   // [80,80]
    float* out = (float*)d_out;
    float* ws  = (float*)d_ws;

    // workspace layout (floats), total ~1.47M floats (~5.9 MB)
    float* T1 = ws;                  // 80*1024
    float* L1 = T1 + 80 * 1024;      // 80*1024
    float* T2 = L1 + 80 * 1024;      // 80*2048
    float* XL = T2 + 80 * 2048;      // 80*2048
    float* T3 = XL + 80 * 2048;      // 80*4096
    float* L3 = T3 + 80 * 4096;      // 80*4096
    float* T4 = L3 + 80 * 4096;      // 80*2048
    float* XG = T4 + 80 * 2048;      // 80*2048

    dim3 blk(256);
    auto grd = [](int N) { return dim3(N / 128, 5); };   // M = 80 = 5 tiles

    // local branch: XL = A1 @ (leaky(A1 @ (inp @ W1)) @ W2)
    gemm_wmma_f16<<<grd(1024), blk, 0, stream>>>(inp, W1, T1, 80,  300, 1024, 0);
    gemm_wmma_f16<<<grd(1024), blk, 0, stream>>>(A1,  T1, L1, 80,   80, 1024, 1);
    gemm_wmma_f16<<<grd(2048), blk, 0, stream>>>(L1,  W2, T2, 80, 1024, 2048, 0);
    gemm_wmma_f16<<<grd(2048), blk, 0, stream>>>(A1,  T2, XL, 80,   80, 2048, 0);

    // global branch: XG = A2 @ (leaky(A2 @ (cor @ W3)) @ W4)
    gemm_wmma_f16<<<grd(4096), blk, 0, stream>>>(cor, W3, T3, 80, 1536, 4096, 0);
    gemm_wmma_f16<<<grd(4096), blk, 0, stream>>>(A2,  T3, L3, 80,   80, 4096, 1);
    gemm_wmma_f16<<<grd(2048), blk, 0, stream>>>(L3,  W4, T4, 80, 4096, 2048, 0);
    gemm_wmma_f16<<<grd(2048), blk, 0, stream>>>(A2,  T4, XG, 80,   80, 2048, 0);

    // streaming broadcast-mul + window-sum (HBM-store-bound phase)
    winsum_kernel<<<dim3(160, 256), blk, 0, stream>>>(feature, XL, XG, out);
}
